// LinearAttention_14319420964902
// MI455X (gfx1250) — matrix operand order repaired and checked by reading, not compiled, not run on hardware
//
#include <hip/hip_runtime.h>

// Linear attention, MI455X (gfx1250, wave32).
// WMMA f32_16x16x32_f16 GEMM cores + TDM (tensor_load_to_lds) for the pure-copy
// B-operand staging in phase 2. B=32, L=4096, D=256, fp32 in/out.

typedef __attribute__((ext_vector_type(16))) _Float16 v16h;
typedef __attribute__((ext_vector_type(8)))  _Float16 v8h;
typedef __attribute__((ext_vector_type(8)))  float    v8f;
typedef unsigned int u32x4 __attribute__((ext_vector_type(4)));
typedef int          i32x8 __attribute__((ext_vector_type(8)));
typedef int          i32x4 __attribute__((ext_vector_type(4)));

#define BATCH 32
#define LSEQ  4096
#define DDIM  256
#define LDSP  40        // padded LDS row stride in halves (80B, 16B-aligned)
#define FEPS  1e-6f

#if defined(__has_builtin)
#  if __has_builtin(__builtin_amdgcn_tensor_load_to_lds) && __has_builtin(__builtin_amdgcn_s_wait_tensorcnt)
#    define HAVE_TDM 1
#  endif
#endif
#ifndef HAVE_TDM
#  define HAVE_TDM 0
#endif

// elu(x)+1 : x>0 ? x+1 : exp(x)
__device__ __forceinline__ float fmap(float x) {
    return x > 0.0f ? x + 1.0f : __expf(x);
}

// ---- WMMA fragment loaders (wave32, 16x16x32 f16) -------------------------
// A fragment 16x32 (MxK), LDS row-major [m][k], row stride S halves.
__device__ __forceinline__ v16h load_a_frag(const _Float16* base, int S) {
    const int lane = threadIdx.x & 31;
    const int m    = lane & 15;
    const int koff = (lane >> 4) << 4;      // 0 or 16
    const _Float16* p = base + m * S + koff;
    v16h a;
#pragma unroll
    for (int c = 0; c < 8; ++c) a[c]     = p[c];
#pragma unroll
    for (int c = 0; c < 8; ++c) a[8 + c] = p[8 + c];
    return a;
}

// B fragment 32x16 (KxN), stored TRANSPOSED in LDS as [n][k], row stride S.
__device__ __forceinline__ v16h load_b_frag(const _Float16* base, int S) {
    const int lane = threadIdx.x & 31;
    const int n    = lane & 15;
    const int koff = (lane >> 4) << 4;      // 0 or 16
    const _Float16* p = base + n * S + koff;
    v16h b;
#pragma unroll
    for (int c = 0; c < 16; ++c) b[c] = p[c];
    return b;
}

// ---- ks[b][d] = sum_l fmap(k[b][l][d]) ------------------------------------
__global__ __launch_bounds__(256)
void k_colsum(const float* __restrict__ kin, float* __restrict__ ks) {
    const int b = blockIdx.x;
    const int d = threadIdx.x;
    const float* kp = kin + (size_t)b * LSEQ * DDIM + d;
    float acc = 0.0f;
    for (int l = 0; l < LSEQ; ++l)
        acc += fmap(kp[(size_t)l * DDIM]);
    ks[b * DDIM + d] = acc;
}

// ---- phase 1: kvT[d][e] = sum_l fmap(k[l][d]) * fmap(v[l][e]) -------------
// Stored to ws as f16 in [b][e][d] layout (== B operand layout for phase 2).
__global__ __launch_bounds__(256)
void phase1_kv(const float* __restrict__ kin, const float* __restrict__ vin,
               _Float16* __restrict__ kvout) {
    const int b  = blockIdx.x;
    const int e0 = blockIdx.y * 128;

    __shared__ _Float16 sKT[DDIM][LDSP];  // k chunk transposed: [d][l]
    __shared__ _Float16 sV[128][LDSP];    // v chunk transposed: [e][l]

    const int t    = threadIdx.x;
    const int wave = t >> 5;
    const int wd   = (wave & 3) * 64;     // d-tile base of this wave
    const int we   = (wave >> 2) * 64;    // e-tile base of this wave

    v8f acc[4][4];
#pragma unroll
    for (int i = 0; i < 4; ++i)
#pragma unroll
        for (int j = 0; j < 4; ++j) acc[i][j] = (v8f)(0.0f);

    const int lrow = t >> 3;              // 0..31 : l within chunk
    const int dcol = (t & 7) * 32;        // 32 d per thread
    const int ecol = (t & 7) * 16;        // 16 e per thread
    const float* kp = kin + (size_t)b * LSEQ * DDIM;
    const float* vp = vin + (size_t)b * LSEQ * DDIM + e0;

    for (int l0 = 0; l0 < LSEQ; l0 += 32) {
        float4 kr[8];
#pragma unroll
        for (int i = 0; i < 8; ++i)
            kr[i] = *(const float4*)(kp + (size_t)(l0 + lrow) * DDIM + dcol + i * 4);
        float4 vr[4];
#pragma unroll
        for (int i = 0; i < 4; ++i)
            vr[i] = *(const float4*)(vp + (size_t)(l0 + lrow) * DDIM + ecol + i * 4);

        __syncthreads();
#pragma unroll
        for (int i = 0; i < 8; ++i) {
            sKT[dcol + i * 4 + 0][lrow] = (_Float16)fmap(kr[i].x);
            sKT[dcol + i * 4 + 1][lrow] = (_Float16)fmap(kr[i].y);
            sKT[dcol + i * 4 + 2][lrow] = (_Float16)fmap(kr[i].z);
            sKT[dcol + i * 4 + 3][lrow] = (_Float16)fmap(kr[i].w);
        }
#pragma unroll
        for (int i = 0; i < 4; ++i) {
            sV[ecol + i * 4 + 0][lrow] = (_Float16)fmap(vr[i].x);
            sV[ecol + i * 4 + 1][lrow] = (_Float16)fmap(vr[i].y);
            sV[ecol + i * 4 + 2][lrow] = (_Float16)fmap(vr[i].z);
            sV[ecol + i * 4 + 3][lrow] = (_Float16)fmap(vr[i].w);
        }
        __syncthreads();

        v16h afr[4];
#pragma unroll
        for (int i = 0; i < 4; ++i)
            afr[i] = load_a_frag(&sKT[wd + i * 16][0], LDSP);
#pragma unroll
        for (int j = 0; j < 4; ++j) {
            v16h bfr = load_b_frag(&sV[we + j * 16][0], LDSP);
#pragma unroll
            for (int i = 0; i < 4; ++i)
                acc[i][j] = __builtin_amdgcn_wmma_f32_16x16x32_f16(
                    false, afr[i], false, bfr, (short)0, acc[i][j], false, false);
        }
    }

    // writeback acc (C layout: N = lane&15, M = r + 8*(lane>>4)) as f16 [b][e][d]
    const int lane = t & 31;
    const int n    = lane & 15;
    const int hi   = lane >> 4;
#pragma unroll
    for (int i = 0; i < 4; ++i)
#pragma unroll
        for (int j = 0; j < 4; ++j) {
            const int e = e0 + we + j * 16 + n;
            const int d = wd + i * 16 + hi * 8;
            v8h h;
#pragma unroll
            for (int c = 0; c < 8; ++c) h[c] = (_Float16)acc[i][j][c];
            *(v8h*)(kvout + ((size_t)b * DDIM + e) * DDIM + d) = h;
        }
}

// ---- phase 2: out[l][e] = (sum_d fmap(q[l][d]) * kvT[d][e]) / (qq.ks + eps)
// grid (B, 32): block covers 128 l x 256 e. 8 waves, 64x64 per wave.
// B operand (kv tile, pure f16 copy) is staged into LDS by the Tensor Data
// Mover: 2-D D#, tile 16 DWORDs x 256 rows, tensor row stride 128 DWORDs,
// LDS padding 4 DWORDs every 16 DWORDs -> 80B LDS pitch (= LDSP halves).
__global__ __launch_bounds__(256)
void phase2_out(const float* __restrict__ qin, const _Float16* __restrict__ kv,
                const float* __restrict__ ks, float* __restrict__ out) {
    const int b  = blockIdx.x;
    const int l0 = blockIdx.y * 128;

    __shared__ _Float16 sA[128][LDSP];    // qq chunk row-major: [l][d]
    __shared__ _Float16 sB[DDIM][LDSP];   // kv chunk: [e][d] (transposed B)
    __shared__ float    sks[32];
    __shared__ float    sz[128];

    const int t    = threadIdx.x;
    const int wave = t >> 5;
    const int wl   = (wave & 1) * 64;     // l-tile base
    const int we   = (wave >> 1) * 64;    // e-tile base

    v8f acc[4][4];
#pragma unroll
    for (int i = 0; i < 4; ++i)
#pragma unroll
        for (int j = 0; j < 4; ++j) acc[i][j] = (v8f)(0.0f);
    float zacc = 0.0f;

    const int lrow = t >> 1;              // 0..127
    const int dcol = (t & 1) * 16;
    const float* qp = qin + ((size_t)b * LSEQ + l0) * DDIM;
    const _Float16* kvbase = kv + (size_t)b * DDIM * DDIM;

#if HAVE_TDM
    const unsigned sB_lds = (unsigned)(unsigned long long)(uintptr_t)&sB[0][0];
#else
    const _Float16* kvp = kvbase + (size_t)t * DDIM;  // row e = t
#endif

    for (int d0 = 0; d0 < DDIM; d0 += 32) {
        float4 qr[4];
#pragma unroll
        for (int i = 0; i < 4; ++i)
            qr[i] = *(const float4*)(qp + (size_t)lrow * DDIM + d0 + dcol + i * 4);
#if !HAVE_TDM
        uint4 br[4];
#pragma unroll
        for (int i = 0; i < 4; ++i)
            br[i] = *(const uint4*)((const char*)kvp + d0 * 2 + i * 16);
#endif
        const float kse = (t < 32) ? ks[b * DDIM + d0 + t] : 0.0f;

        __syncthreads();

#if HAVE_TDM
        if (wave == 0) {
            // D# group 0: count=1 | lds_addr | global_addr(57b) | type=2
            const unsigned long long ga =
                (unsigned long long)(uintptr_t)kvbase + (unsigned)(d0 * 2);
            u32x4 g0;
            g0[0] = 1u;                                    // count=1 (user D#)
            g0[1] = sB_lds;                                // LDS byte address
            g0[2] = (unsigned)(ga & 0xFFFFFFFFu);          // global_addr[31:0]
            g0[3] = (unsigned)((ga >> 32) & 0x01FFFFFFu)   // global_addr[56:32]
                  | (2u << 30);                            // type=2 (image)
            // D# group 1: data_size=4B, pad_enable, pad_interval=16DW,
            // pad_amount=4DW; tensor 128x256 DW, tile 16x256 DW.
            i32x8 g1;
            g1[0] = (2 << 16) | (1 << 20) | (3 << 22) | (3 << 25);
            g1[1] = (int)(128u << 16);   // tensor_dim0 = 128 DWORDs (low16)
            g1[2] = (int)(256u << 16);   // tensor_dim0 hi=0 ; tensor_dim1 = 256
            g1[3] = (int)(16u << 16);    // tensor_dim1 hi=0 ; tile_dim0 = 16 DW
            g1[4] = 256;                 // tile_dim1 = 256 rows ; tile_dim2 = 0
            g1[5] = 128;                 // tensor_dim0_stride = 128 DW (low32)
            g1[6] = 0;                   // stride hi ; tensor_dim1_stride = 0
            g1[7] = 0;
            const i32x4 gz = {0, 0, 0, 0};
#  if __clang_major__ >= 23
            const i32x8 gz8 = {0, 0, 0, 0, 0, 0, 0, 0};
            __builtin_amdgcn_tensor_load_to_lds(g0, g1, gz, gz, gz8, 0);
#  else
            __builtin_amdgcn_tensor_load_to_lds(g0, g1, gz, gz, 0);
#  endif
        }
#endif

#pragma unroll
        for (int i = 0; i < 4; ++i) {
            sA[lrow][dcol + i * 4 + 0] = (_Float16)fmap(qr[i].x);
            sA[lrow][dcol + i * 4 + 1] = (_Float16)fmap(qr[i].y);
            sA[lrow][dcol + i * 4 + 2] = (_Float16)fmap(qr[i].z);
            sA[lrow][dcol + i * 4 + 3] = (_Float16)fmap(qr[i].w);
        }
#if !HAVE_TDM
#pragma unroll
        for (int i = 0; i < 4; ++i)
            *(uint4*)((char*)&sB[t][0] + i * 16) = br[i];
#endif
        if (t < 32) sks[t] = kse;

#if HAVE_TDM
        if (wave == 0) __builtin_amdgcn_s_wait_tensorcnt(0);
#endif
        __syncthreads();

        // Z partial: threads 0..127 each own one local l row.
        if (t < 128) {
#pragma unroll
            for (int dd = 0; dd < 32; ++dd)
                zacc += (float)sA[t][dd] * sks[dd];
        }

        v16h afr[4];
#pragma unroll
        for (int i = 0; i < 4; ++i)
            afr[i] = load_a_frag(&sA[wl + i * 16][0], LDSP);
#pragma unroll
        for (int j = 0; j < 4; ++j) {
            v16h bfr = load_b_frag(&sB[we + j * 16][0], LDSP);
#pragma unroll
            for (int i = 0; i < 4; ++i)
                acc[i][j] = __builtin_amdgcn_wmma_f32_16x16x32_f16(
                    false, afr[i], false, bfr, (short)0, acc[i][j], false, false);
        }
    }

    if (t < 128) sz[t] = 1.0f / (zacc + FEPS);
    __syncthreads();

    const int lane = t & 31;
    const int n    = lane & 15;
    const int hi   = lane >> 4;
#pragma unroll
    for (int i = 0; i < 4; ++i)
#pragma unroll
        for (int j = 0; j < 4; ++j) {
            const int e  = we + j * 16 + n;
            const int lb = wl + i * 16 + hi * 8;
#pragma unroll
            for (int r = 0; r < 8; ++r) {
                const int ll = lb + r;
                out[((size_t)b * LSEQ + l0 + ll) * DDIM + e] = acc[i][j][r] * sz[ll];
            }
        }
}

extern "C" void kernel_launch(void* const* d_in, const int* in_sizes, int n_in,
                              void* d_out, int out_size, void* d_ws, size_t ws_size,
                              hipStream_t stream) {
    (void)in_sizes; (void)n_in; (void)out_size; (void)ws_size;
    const float* q = (const float*)d_in[0];
    const float* k = (const float*)d_in[1];
    const float* v = (const float*)d_in[2];
    float* out = (float*)d_out;

    // workspace: kv f16 [B][e][d] then ks f32 [B][d]
    _Float16* ws_kv = (_Float16*)d_ws;
    float*    ws_ks = (float*)((char*)d_ws + (size_t)BATCH * DDIM * DDIM * 2);

    k_colsum<<<dim3(BATCH), dim3(256), 0, stream>>>(k, ws_ks);
    phase1_kv<<<dim3(BATCH, 2), dim3(256), 0, stream>>>(k, v, ws_kv);
    phase2_out<<<dim3(BATCH, LSEQ / 128), dim3(256), 0, stream>>>(q, ws_kv, ws_ks, out);
}
